// TNTDecoder_79697413145182
// MI455X (gfx1250) — matrix operand order repaired and checked by reading, not compiled, not run on hardware
//
#include <hip/hip_runtime.h>
#include <hip/hip_bf16.h>

#define C_DIM 128
#define H_DIM 64
#define N_CAND 2000
#define M_SEL 50
#define HOR 30
#define LN_EPS 1e-5f

typedef __attribute__((ext_vector_type(16))) _Float16 v16h;
typedef __attribute__((ext_vector_type(8)))  _Float16 v8h;
typedef __attribute__((ext_vector_type(4)))  _Float16 v4h;
typedef __attribute__((ext_vector_type(8)))  float    v8f;
typedef __attribute__((ext_vector_type(4)))  int      v4i;

#define GLOBAL_AS __attribute__((address_space(1)))
#define LDS_AS    __attribute__((address_space(3)))

#if defined(__gfx1250__) && __has_builtin(__builtin_amdgcn_global_load_async_to_lds_b128)
#define HAVE_ASYNC_LDS 1
#else
#define HAVE_ASYNC_LDS 0
#endif

__device__ __forceinline__ void wait_asynccnt0() {
#if defined(__gfx1250__)
#if __has_builtin(__builtin_amdgcn_s_wait_asynccnt)
    __builtin_amdgcn_s_wait_asynccnt(0);
#else
    asm volatile("s_wait_asynccnt 0" ::: "memory");
#endif
#endif
}

// ---------------------------------------------------------------------------
// Stage 0a: featproj[b][j] = b1[j] + sum_k feat[b][k] * w1[k][j]   (k<128)
// ---------------------------------------------------------------------------
__global__ void tnt_featproj_kernel(const float* __restrict__ feat,
                                    const float* __restrict__ w1,
                                    const float* __restrict__ b1,
                                    float* __restrict__ outp) {
    const int b = blockIdx.x, j = threadIdx.x;   // blockDim = 64
    float s = b1[j];
    const float* f = feat + (size_t)b * C_DIM;
#pragma unroll 8
    for (int k = 0; k < C_DIM; ++k) s = fmaf(f[k], w1[k * H_DIM + j], s);
    outp[b * H_DIM + j] = s;
}

// Stage 0b: fp32 [64,64] (k,n) -> fp16 TRANSPOSED [64,64] (n,k) for B fragments
__global__ void tnt_cvt_w2t_kernel(const float* __restrict__ src,
                                   _Float16* __restrict__ dstT) {
    int i = blockIdx.x * blockDim.x + threadIdx.x;   // 4096 elements
    if (i < H_DIM * H_DIM) {
        const int k = i >> 6, n = i & 63;
        dstT[n * H_DIM + k] = (_Float16)src[i];
    }
}

// ---------------------------------------------------------------------------
// Stage 1: fused MLP tail over a 64-row tile. blockDim = 128 (4 waves).
//   h0 = featproj + x*w1[128] + y*w1[129] -> LN -> ReLU -> f16
//   h2 = h1 @ W2 (v_wmma_f32_16x16x32_f16) + b2 -> LN -> ReLU
//   out = h2 @ W3 + b3 (+ optional per-row add)
// ---------------------------------------------------------------------------
#define H0_S 68   // float stride (272B, 16B aligned rows)
#define H1_S 72   // half stride  (144B, 16B aligned rows)

__global__ __launch_bounds__(128)
void tnt_mlp_tail_kernel(const float* __restrict__ featproj,   // [B,64] (incl b1)
                         const float* __restrict__ w1xy,       // [2,64]
                         const float* __restrict__ g1,
                         const float* __restrict__ be1,
                         const _Float16* __restrict__ w2hT,    // [64,64] (n,k)
                         const float* __restrict__ b2,
                         const float* __restrict__ g2,
                         const float* __restrict__ be2,
                         const float* __restrict__ w3,         // [64,outDim]
                         const float* __restrict__ b3,         // [outDim]
                         const float* __restrict__ xy,         // [B*rowsPerBatch,2]
                         const float* __restrict__ addv,       // optional
                         float* __restrict__ out,              // [B*rowsPerBatch,outDim]
                         int rowsPerBatch, int outDim) {
    __shared__ __align__(16) _Float16 sW2T[64][H1_S];   // W2 transposed (n,k)
    __shared__ __align__(16) float    sH0[64][H0_S];    // h0, later h2
    __shared__ __align__(16) _Float16 sH1[64][H1_S];    // h1 (f16)
    __shared__ __align__(16) float sFP[64], sW1X[64], sW1Y[64], sG1[64], sBE1[64];
    __shared__ __align__(16) float sB2[64], sG2[64], sBE2[64];
    __shared__ __align__(16) float sX[64], sY[64], sMu[64], sRs[64];

    const int tid = threadIdx.x;
    const int b = blockIdx.y;
    const int rowBase = blockIdx.x * 64;

    // ---- W2T -> LDS ----
#if HAVE_ASYNC_LDS
    // async DMA: 512 x 16B, single ASYNCcnt wait, no VGPR round trip
    for (int i = tid; i < 512; i += 128) {
        const int r = i >> 3, c = (i & 7) * 8;    // 8 halfs = 16B per op
        __builtin_amdgcn_global_load_async_to_lds_b128(
            (GLOBAL_AS v4i*)(w2hT + i * 8),
            (LDS_AS v4i*)&sW2T[r][c], 0, 0);
    }
#else
    for (int i = tid; i < 1024; i += 128) {
        const int r = i >> 4, c = (i & 15) * 4;
        *(uint2*)&sW2T[r][c] = ((const uint2*)w2hT)[i];
    }
#endif

    if (tid < 64) {
        sFP[tid]  = featproj[b * 64 + tid];
        sW1X[tid] = w1xy[tid];
        sW1Y[tid] = w1xy[64 + tid];
        sG1[tid]  = g1[tid];  sBE1[tid] = be1[tid];
        sB2[tid]  = b2[tid];  sG2[tid]  = g2[tid];  sBE2[tid] = be2[tid];
        const int rrow = rowBase + tid;
        const bool valid = rrow < rowsPerBatch;
        const long gr = (long)b * rowsPerBatch + rrow;
        sX[tid] = valid ? xy[gr * 2 + 0] : 0.f;
        sY[tid] = valid ? xy[gr * 2 + 1] : 0.f;
    }
#if HAVE_ASYNC_LDS
    wait_asynccnt0();
#endif
    __syncthreads();

    // ---- layer 1 elementwise part (float4) ----
    for (int idx = tid; idx < 64 * 16; idx += 128) {
        const int r = idx >> 4, j = (idx & 15) * 4;
        const float4 fp = *(const float4*)&sFP[j];
        const float4 wx = *(const float4*)&sW1X[j];
        const float4 wy = *(const float4*)&sW1Y[j];
        const float x = sX[r], y = sY[r];
        float4 h;
        h.x = fmaf(y, wy.x, fmaf(x, wx.x, fp.x));
        h.y = fmaf(y, wy.y, fmaf(x, wx.y, fp.y));
        h.z = fmaf(y, wy.z, fmaf(x, wx.z, fp.z));
        h.w = fmaf(y, wy.w, fmaf(x, wx.w, fp.w));
        *(float4*)&sH0[r][j] = h;
    }
    __syncthreads();

    // ---- LN1 stats (float4 row scans) ----
    if (tid < 64) {
        const float4* row = (const float4*)&sH0[tid][0];
        float4 a = {0.f, 0.f, 0.f, 0.f};
#pragma unroll
        for (int q = 0; q < 16; ++q) {
            const float4 v = row[q];
            a.x += v.x; a.y += v.y; a.z += v.z; a.w += v.w;
        }
        const float mu = (a.x + a.y + a.z + a.w) * (1.f / 64.f);
        float4 s = {0.f, 0.f, 0.f, 0.f};
#pragma unroll
        for (int q = 0; q < 16; ++q) {
            const float4 v = row[q];
            float d;
            d = v.x - mu; s.x = fmaf(d, d, s.x);
            d = v.y - mu; s.y = fmaf(d, d, s.y);
            d = v.z - mu; s.z = fmaf(d, d, s.z);
            d = v.w - mu; s.w = fmaf(d, d, s.w);
        }
        const float var = (s.x + s.y + s.z + s.w) * (1.f / 64.f);
        sMu[tid] = mu;
        sRs[tid] = rsqrtf(var + LN_EPS);
    }
    __syncthreads();

    // ---- LN1 apply + ReLU -> f16 (float4 in, v4h out) ----
    for (int idx = tid; idx < 64 * 16; idx += 128) {
        const int r = idx >> 4, j = (idx & 15) * 4;
        const float4 v = *(const float4*)&sH0[r][j];
        const float4 g = *(const float4*)&sG1[j];
        const float4 be = *(const float4*)&sBE1[j];
        const float mu = sMu[r], rs = sRs[r];
        v4h st;
        st[0] = (_Float16)fmaxf(fmaf((v.x - mu) * rs, g.x, be.x), 0.f);
        st[1] = (_Float16)fmaxf(fmaf((v.y - mu) * rs, g.y, be.y), 0.f);
        st[2] = (_Float16)fmaxf(fmaf((v.z - mu) * rs, g.z, be.z), 0.f);
        st[3] = (_Float16)fmaxf(fmaf((v.w - mu) * rs, g.w, be.w), 0.f);
        *(v4h*)&sH1[r][j] = st;
    }
    __syncthreads();

    // ---- mid layer: h1[16x64] @ W2[64x64] via WMMA (per wave) ----
    const int wave = tid >> 5;
    const int lane = tid & 31;
    const int mrow = lane & 15;
    const int hi   = lane >> 4;
    const int arow = wave * 16 + mrow;

    // A fragments: contiguous b128 pieces of the h1 row
    const v8h a0lo = *(const v8h*)&sH1[arow][hi * 16];
    const v8h a0hi = *(const v8h*)&sH1[arow][hi * 16 + 8];
    const v8h a1lo = *(const v8h*)&sH1[arow][32 + hi * 16];
    const v8h a1hi = *(const v8h*)&sH1[arow][32 + hi * 16 + 8];
    const v16h a0 = __builtin_shufflevector(a0lo, a0hi, 0,1,2,3,4,5,6,7,8,9,10,11,12,13,14,15);
    const v16h a1 = __builtin_shufflevector(a1lo, a1hi, 0,1,2,3,4,5,6,7,8,9,10,11,12,13,14,15);

#pragma unroll
    for (int nt = 0; nt < 4; ++nt) {
        const int nn = nt * 16 + mrow;
        const int k0 = hi * 16;
        const v8h b0lo = *(const v8h*)&sW2T[nn][k0];
        const v8h b0hi = *(const v8h*)&sW2T[nn][k0 + 8];
        const v8h b1lo = *(const v8h*)&sW2T[nn][32 + k0];
        const v8h b1hi = *(const v8h*)&sW2T[nn][32 + k0 + 8];
        const v16h bf0 = __builtin_shufflevector(b0lo, b0hi, 0,1,2,3,4,5,6,7,8,9,10,11,12,13,14,15);
        const v16h bf1 = __builtin_shufflevector(b1lo, b1hi, 0,1,2,3,4,5,6,7,8,9,10,11,12,13,14,15);
        v8f acc = {};
        acc = __builtin_amdgcn_wmma_f32_16x16x32_f16(false, a0, false, bf0,
                                                     (short)0, acc, false, false);
        acc = __builtin_amdgcn_wmma_f32_16x16x32_f16(false, a1, false, bf1,
                                                     (short)0, acc, false, false);
#pragma unroll
        for (int i = 0; i < 8; ++i) {
            const int rr = wave * 16 + hi * 8 + i;
            sH0[rr][nn] = acc[i] + sB2[nn];      // h2 = h1@W2 + b2
        }
    }
    __syncthreads();

    // ---- LN2 stats ----
    if (tid < 64) {
        const float4* row = (const float4*)&sH0[tid][0];
        float4 a = {0.f, 0.f, 0.f, 0.f};
#pragma unroll
        for (int q = 0; q < 16; ++q) {
            const float4 v = row[q];
            a.x += v.x; a.y += v.y; a.z += v.z; a.w += v.w;
        }
        const float mu = (a.x + a.y + a.z + a.w) * (1.f / 64.f);
        float4 s = {0.f, 0.f, 0.f, 0.f};
#pragma unroll
        for (int q = 0; q < 16; ++q) {
            const float4 v = row[q];
            float d;
            d = v.x - mu; s.x = fmaf(d, d, s.x);
            d = v.y - mu; s.y = fmaf(d, d, s.y);
            d = v.z - mu; s.z = fmaf(d, d, s.z);
            d = v.w - mu; s.w = fmaf(d, d, s.w);
        }
        const float var = (s.x + s.y + s.z + s.w) * (1.f / 64.f);
        sMu[tid] = mu;
        sRs[tid] = rsqrtf(var + LN_EPS);
    }
    __syncthreads();

    // ---- LN2 apply + ReLU (in place, float4) ----
    for (int idx = tid; idx < 64 * 16; idx += 128) {
        const int r = idx >> 4, j = (idx & 15) * 4;
        float4 v = *(const float4*)&sH0[r][j];
        const float4 g = *(const float4*)&sG2[j];
        const float4 be = *(const float4*)&sBE2[j];
        const float mu = sMu[r], rs = sRs[r];
        v.x = fmaxf(fmaf((v.x - mu) * rs, g.x, be.x), 0.f);
        v.y = fmaxf(fmaf((v.y - mu) * rs, g.y, be.y), 0.f);
        v.z = fmaxf(fmaf((v.z - mu) * rs, g.z, be.z), 0.f);
        v.w = fmaxf(fmaf((v.w - mu) * rs, g.w, be.w), 0.f);
        *(float4*)&sH0[r][j] = v;
    }
    __syncthreads();

    // ---- head layer ----
    for (int idx = tid; idx < 64 * outDim; idx += 128) {
        const int r = idx / outDim;
        const int o = idx - r * outDim;
        const int rrow = rowBase + r;
        if (rrow < rowsPerBatch) {
            float s = b3[o];
            const float4* hrow = (const float4*)&sH0[r][0];
#pragma unroll 4
            for (int q = 0; q < 16; ++q) {
                const float4 h4 = hrow[q];
                const int jb = q * 4;
                s = fmaf(h4.x, w3[(jb + 0) * outDim + o], s);
                s = fmaf(h4.y, w3[(jb + 1) * outDim + o], s);
                s = fmaf(h4.z, w3[(jb + 2) * outDim + o], s);
                s = fmaf(h4.w, w3[(jb + 3) * outDim + o], s);
            }
            const long gr = (long)b * rowsPerBatch + rrow;
            if (addv) s += addv[gr];
            out[gr * outDim + o] = s;
        }
    }
}

// ---------------------------------------------------------------------------
// Stage 2: softmax over N + iterative top-50 + gather, one block per batch b.
// ---------------------------------------------------------------------------
__global__ __launch_bounds__(256)
void tnt_topk_kernel(const float* __restrict__ logits,   // [B,N]
                     const float* __restrict__ tarc,     // [B,N,2]
                     const float* __restrict__ offs,     // [B,N,2]
                     float* __restrict__ probOut,        // [B,M]
                     float* __restrict__ selXY) {        // [B,M,2]
    __shared__ float sLog[N_CAND];
    __shared__ float sV[256];
    __shared__ int   sI[256];
    __shared__ float sMax, sSum;
    const int b = blockIdx.x, tid = threadIdx.x;

    for (int i = tid; i < N_CAND; i += 256) sLog[i] = logits[(size_t)b * N_CAND + i];
    __syncthreads();

    float mv = -3.4e38f;
    for (int i = tid; i < N_CAND; i += 256) mv = fmaxf(mv, sLog[i]);
    sV[tid] = mv; __syncthreads();
    for (int s = 128; s > 0; s >>= 1) {
        if (tid < s) sV[tid] = fmaxf(sV[tid], sV[tid + s]);
        __syncthreads();
    }
    if (tid == 0) sMax = sV[0];
    __syncthreads();
    const float gmax = sMax;

    float acc = 0.f;
    for (int i = tid; i < N_CAND; i += 256) acc += __expf(sLog[i] - gmax);
    sV[tid] = acc; __syncthreads();
    for (int s = 128; s > 0; s >>= 1) {
        if (tid < s) sV[tid] += sV[tid + s];
        __syncthreads();
    }
    if (tid == 0) sSum = sV[0];
    __syncthreads();
    const float invSum = 1.f / sSum;

    for (int m = 0; m < M_SEL; ++m) {
        float bv = -3.4e38f; int bi = N_CAND;
        for (int i = tid; i < N_CAND; i += 256) {
            const float v = sLog[i];
            if (v > bv || (v == bv && i < bi)) { bv = v; bi = i; }
        }
        sV[tid] = bv; sI[tid] = bi; __syncthreads();
        for (int s = 128; s > 0; s >>= 1) {
            if (tid < s) {
                const float ov = sV[tid + s]; const int oi = sI[tid + s];
                if (ov > sV[tid] || (ov == sV[tid] && oi < sI[tid])) {
                    sV[tid] = ov; sI[tid] = oi;
                }
            }
            __syncthreads();
        }
        if (tid == 0) {
            const int idx = sI[0]; const float val = sV[0];
            probOut[b * M_SEL + m] = __expf(val - gmax) * invSum;
            const size_t gi = (size_t)b * N_CAND + idx;
            selXY[((size_t)b * M_SEL + m) * 2 + 0] = tarc[gi * 2 + 0] + offs[gi * 2 + 0];
            selXY[((size_t)b * M_SEL + m) * 2 + 1] = tarc[gi * 2 + 1] + offs[gi * 2 + 1];
            sLog[idx] = -3.4e38f;
        }
        __syncthreads();
    }
}

// ---------------------------------------------------------------------------
extern "C" void kernel_launch(void* const* d_in, const int* in_sizes, int n_in,
                              void* d_out, int out_size, void* d_ws, size_t ws_size,
                              hipStream_t stream) {
    const float* feat = (const float*)d_in[0];   // [B,1,C]
    const float* tarc = (const float*)d_in[1];   // [B,N,2]
    const float* mask = (const float*)d_in[2];   // [B,N,1]
    // params in insertion order: prob, offset, traj; each {w1,b1,g1,be1,w2,b2,g2,be2,w3,b3}
    const float* P[3][10];
    for (int mi = 0; mi < 3; ++mi)
        for (int k = 0; k < 10; ++k) P[mi][k] = (const float*)d_in[3 + mi * 10 + k];

    const int B = in_sizes[0] / C_DIM;

    char* ws = (char*)d_ws;
    size_t off = 0;
    float* fp[3]; _Float16* w2hT[3];
    for (int i = 0; i < 3; ++i) { fp[i] = (float*)(ws + off); off += (size_t)B * H_DIM * 4; }
    for (int i = 0; i < 3; ++i) { w2hT[i] = (_Float16*)(ws + off); off += H_DIM * H_DIM * 2; }
    off = (off + 255) & ~(size_t)255;
    float* logits = (float*)(ws + off); off += (size_t)B * N_CAND * 4;
    float* offs   = (float*)(ws + off); off += (size_t)B * N_CAND * 2 * 4;
    float* selXY  = (float*)(ws + off); off += (size_t)B * M_SEL * 2 * 4;

    for (int i = 0; i < 3; ++i) {
        tnt_featproj_kernel<<<dim3(B), dim3(H_DIM), 0, stream>>>(feat, P[i][0], P[i][1], fp[i]);
        tnt_cvt_w2t_kernel<<<dim3((H_DIM * H_DIM + 255) / 256), dim3(256), 0, stream>>>(
            P[i][4], w2hT[i]);
    }

    const int tilesN = (N_CAND + 63) / 64;
    // prob MLP -> logits (+mask folded in)
    tnt_mlp_tail_kernel<<<dim3(tilesN, B), dim3(128), 0, stream>>>(
        fp[0], P[0][0] + C_DIM * H_DIM, P[0][2], P[0][3], w2hT[0],
        P[0][5], P[0][6], P[0][7], P[0][8], P[0][9],
        tarc, mask, logits, N_CAND, 1);
    // offset MLP -> offsets
    tnt_mlp_tail_kernel<<<dim3(tilesN, B), dim3(128), 0, stream>>>(
        fp[1], P[1][0] + C_DIM * H_DIM, P[1][2], P[1][3], w2hT[1],
        P[1][5], P[1][6], P[1][7], P[1][8], P[1][9],
        tarc, nullptr, offs, N_CAND, 2);

    float* outTraj = (float*)d_out;                          // [B,M,60]
    float* outProb = outTraj + (size_t)B * M_SEL * HOR * 2;  // [B,M,1]
    tnt_topk_kernel<<<dim3(B), dim3(256), 0, stream>>>(logits, tarc, offs, outProb, selXY);

    // traj MLP on selected targets -> d_out
    tnt_mlp_tail_kernel<<<dim3(1, B), dim3(128), 0, stream>>>(
        fp[2], P[2][0] + C_DIM * H_DIM, P[2][2], P[2][3], w2hT[2],
        P[2][5], P[2][6], P[2][7], P[2][8], P[2][9],
        selXY, nullptr, outTraj, M_SEL, HOR * 2);
}
